// MoEFF_7404523618551
// MI455X (gfx1250) — compile-verified, matrix operand order checked by reading, not run on hardware
//
#include <hip/hip_runtime.h>
#include <cstdint>
#include <cstddef>

// ---------------- problem constants ----------------
#define T_TOK 4096     // B*S
#define DMODEL 1024
#define HEXP 1024
#define SHID 2048
#define NEXP 8
#define TOPK 4

// ---------------- GEMM tiling ----------------
#define BM 128
#define BN 128
#define BK 32
#define LDT 40    // halves per LDS tile row (32 + 8 pad) -> 80B stride: 16B aligned & conflict-free

typedef __attribute__((ext_vector_type(16))) __bf16 v16bf;
typedef __attribute__((ext_vector_type(8)))  float  v8f;
typedef __attribute__((ext_vector_type(4)))  int    v4i;
typedef unsigned short u16;
typedef unsigned int   u32;

// ----- optional CDNA5 async global->LDS path (compile-time guarded) -----
#if defined(__has_builtin)
#if __has_builtin(__builtin_amdgcn_global_load_async_to_lds_b128) && \
    __has_builtin(__builtin_amdgcn_s_wait_asynccnt)
#define USE_ASYNC_LDS 1
#endif
#endif
#ifndef USE_ASYNC_LDS
#define USE_ASYNC_LDS 0
#endif

__device__ __forceinline__ u16 f2bf(float f) {
  u32 u = __builtin_bit_cast(u32, f);
  u32 r = (u + 0x7FFFu + ((u >> 16) & 1u)) >> 16;   // round-to-nearest-even
  return (u16)r;
}
__device__ __forceinline__ float bf2f(u16 h) {
  u32 u = ((u32)h) << 16;
  return __builtin_bit_cast(float, u);
}
__device__ __forceinline__ u32 pack2(float a, float b) {
  return (u32)f2bf(a) | ((u32)f2bf(b) << 16);
}
__device__ __forceinline__ float silu(float x) {
  return x * (1.0f / (1.0f + __expf(-x)));
}

#if USE_ASYNC_LDS
typedef __attribute__((address_space(1))) v4i gv4i;   // global (prints as __device__)
typedef __attribute__((address_space(3))) v4i lv4i;   // LDS (prints as __shared__)
__device__ __forceinline__ void async_copy16(const u16* g, u16* l) {
  __builtin_amdgcn_global_load_async_to_lds_b128((gv4i*)g, (lv4i*)l, 0, 0);
}
#endif

union BF16Frag { u32 u[8]; u16 h[16]; v16bf v; };

// Operand fragment (16 rows x 32 K), tile stored row-major [row][k], stride LDT halves.
// Lane l: row = base + (l&15), k-base = (l>>4)*8; halves j<8 -> k=kb+j, j>=8 -> k=kb+16+(j-8).
// 80B row stride is 16B-aligned -> lowers to 2x ds_load_b128 per fragment.
__device__ __forceinline__ v16bf load_frag(const u16* tile, int base, int lane) {
  const u16* p = tile + (size_t)(base + (lane & 15)) * LDT + ((lane >> 4) << 3);
  BF16Frag f;
#pragma unroll
  for (int t = 0; t < 4; ++t) {
    f.u[t]     = *(const u32*)(p + 2 * t);
    f.u[4 + t] = *(const u32*)(p + 16 + 2 * t);
  }
  return f.v;
}

// C = A(bf16,[M,K]) * Bt(bf16,[N,K])^T (+ bias)
// EPI 0: store bf16 (+bias). EPI 1: store f32 (+bias). EPI 2: f32 Dst += rowscale[m]*acc.
template <int EPI>
__global__ __launch_bounds__(256)
void gemm_bf16_wmma(const u16* __restrict__ A, const u16* __restrict__ Bt,
                    const float* __restrict__ bias, void* __restrict__ Dst,
                    const float* __restrict__ rowscale, int rsStride,
                    int M, int N, int Kd) {
  __shared__ u16 As[2][BM * LDT];
  __shared__ u16 Bs[2][BN * LDT];

  const int tid  = threadIdx.x;
  const int lane = tid & 31;
  const int wave = tid >> 5;
  const int wm   = (wave & 1) * 64;   // 2 waves tile M
  const int wn   = (wave >> 1) * 32;  // 4 waves tile N
  const int bm   = blockIdx.y * BM;
  const int bn   = blockIdx.x * BN;

  // stage-in: both tiles are 128 rows x 32 halves; thread -> (row, 16-half chunk)
  const int sr = tid >> 1;        // row 0..127
  const int sc = (tid & 1) * 16;  // col 0 / 16

  const u16* gA = A  + (size_t)(bm + sr) * Kd + sc;
  const u16* gB = Bt + (size_t)(bn + sr) * Kd + sc;

  v8f acc[4][2] = {};
  const int nk = Kd / BK;

  auto stage = [&](int kt, int buf) {
    const u16* ga = gA + (size_t)kt * BK;
    const u16* gb = gB + (size_t)kt * BK;
    u16* la = &As[buf][sr * LDT + sc];
    u16* lb = &Bs[buf][sr * LDT + sc];
#if USE_ASYNC_LDS
    async_copy16(ga,     la);
    async_copy16(ga + 8, la + 8);
    async_copy16(gb,     lb);
    async_copy16(gb + 8, lb + 8);
#else
    uint4 a0 = *(const uint4*)ga;
    uint4 a1 = *(const uint4*)(ga + 8);
    uint4 b0 = *(const uint4*)gb;
    uint4 b1 = *(const uint4*)(gb + 8);
    *(uint4*)(la + 0) = a0;
    *(uint4*)(la + 8) = a1;
    *(uint4*)(lb + 0) = b0;
    *(uint4*)(lb + 8) = b1;
#endif
    if (kt + 1 < nk) {  // hint next K-tile (global_prefetch_b8)
      __builtin_prefetch(ga + BK, 0, 1);
      __builtin_prefetch(gb + BK, 0, 1);
    }
  };

  stage(0, 0);
  for (int kt = 0; kt < nk; ++kt) {
#if USE_ASYNC_LDS
    __builtin_amdgcn_s_wait_asynccnt(0);  // my async LDS writes landed
#endif
    __syncthreads();                      // everyone's writes visible
    const int buf = kt & 1;
    if (kt + 1 < nk) stage(kt + 1, buf ^ 1);

    v16bf af[4], bfr[2];
#pragma unroll
    for (int mi = 0; mi < 4; ++mi) af[mi] = load_frag(As[buf], wm + mi * 16, lane);
#pragma unroll
    for (int ni = 0; ni < 2; ++ni) bfr[ni] = load_frag(Bs[buf], wn + ni * 16, lane);

#pragma unroll
    for (int mi = 0; mi < 4; ++mi)
#pragma unroll
      for (int ni = 0; ni < 2; ++ni)
        acc[mi][ni] = __builtin_amdgcn_wmma_f32_16x16x32_bf16(
            false, af[mi], false, bfr[ni], (short)0, acc[mi][ni], false, false);
  }

  // Epilogue. C/D layout: lane<16 -> M=r, lane>=16 -> M=r+8; N = lane&15.
  const int mhi = (lane >> 4) * 8;
#pragma unroll
  for (int mi = 0; mi < 4; ++mi) {
#pragma unroll
    for (int ni = 0; ni < 2; ++ni) {
      const int n = bn + wn + ni * 16 + (lane & 15);
      float badd = 0.0f;
      if (EPI != 2 && bias != nullptr) badd = bias[n];
#pragma unroll
      for (int r = 0; r < 8; ++r) {
        const int m = bm + wm + mi * 16 + mhi + r;
        const float v = acc[mi][ni][r] + badd;
        if (EPI == 0) {
          ((u16*)Dst)[(size_t)m * N + n] = f2bf(v);
        } else if (EPI == 1) {
          ((float*)Dst)[(size_t)m * N + n] = v;
        } else {
          float* Df = (float*)Dst;
          const size_t idx = (size_t)m * N + n;
          Df[idx] += rowscale[(size_t)m * rsStride] * v;
        }
      }
    }
  }
}

// ---------------- weight transpose + cast: Wt[n*K+k] = bf16(W[k*N+n]) ----------------
__global__ __launch_bounds__(256)
void transpose_cast_k(const float* __restrict__ W, u16* __restrict__ Wt, int K, int N) {
  __shared__ float tile[32][33];
  const int k0 = blockIdx.y * 32;
  const int n0 = blockIdx.x * 32;
  const int tx = threadIdx.x, ty = threadIdx.y;
#pragma unroll
  for (int i = 0; i < 32; i += 8)
    tile[ty + i][tx] = W[(size_t)(k0 + ty + i) * N + (n0 + tx)];
  __syncthreads();
#pragma unroll
  for (int i = 0; i < 32; i += 8)
    Wt[(size_t)(n0 + ty + i) * K + (k0 + tx)] = f2bf(tile[tx][ty + i]);
}

// ---------------- router: softmax -> greedy top-4 -> renormalize ----------------
__global__ __launch_bounds__(256)
void router_kernel(const u16* __restrict__ h, const float* __restrict__ gate,
                   float* __restrict__ combine) {
  const int gid  = blockIdx.x * blockDim.x + threadIdx.x;
  const int t    = gid >> 5;
  const int lane = gid & 31;
  if (t >= T_TOK) return;

  float s[NEXP] = {};
  for (int d = lane; d < DMODEL; d += 32) {
    const float xv = bf2f(h[(size_t)t * DMODEL + d]);
#pragma unroll
    for (int e = 0; e < NEXP; ++e) s[e] += xv * gate[(size_t)e * DMODEL + d];
  }
#pragma unroll
  for (int off = 16; off > 0; off >>= 1)
#pragma unroll
    for (int e = 0; e < NEXP; ++e) s[e] += __shfl_down(s[e], off, 32);

  if (lane == 0) {
    float mx = s[0];
#pragma unroll
    for (int e = 1; e < NEXP; ++e) mx = fmaxf(mx, s[e]);
    float p[NEXP], sum = 0.0f;
#pragma unroll
    for (int e = 0; e < NEXP; ++e) { p[e] = __expf(s[e] - mx); sum += p[e]; }
#pragma unroll
    for (int e = 0; e < NEXP; ++e) p[e] /= sum;

    float w[NEXP] = {};
    bool used[NEXP] = {};
    float tsum = 0.0f;
#pragma unroll
    for (int k = 0; k < TOPK; ++k) {
      int best = 0; float bv = -1.0f;
#pragma unroll
      for (int e = 0; e < NEXP; ++e)
        if (!used[e] && p[e] > bv) { bv = p[e]; best = e; }
      used[best] = true; w[best] = bv; tsum += bv;
    }
#pragma unroll
    for (int e = 0; e < NEXP; ++e) combine[(size_t)t * NEXP + e] = w[e] / tsum;
  }
}

// ---------------- elementwise kernels (4-wide) ----------------
__global__ void cast_f32_bf16_k(const float* __restrict__ s, u16* __restrict__ d, int n) {
  const int stride = gridDim.x * blockDim.x * 4;
  for (int j = (blockIdx.x * blockDim.x + threadIdx.x) * 4; j < n; j += stride) {
    float4 f = *(const float4*)(s + j);
    *(uint2*)(d + j) = make_uint2(pack2(f.x, f.y), pack2(f.z, f.w));
  }
}

__global__ void swiglu_mul_bf16_k(const u16* __restrict__ a, const u16* __restrict__ b,
                                  u16* __restrict__ d, int n) {
  const int stride = gridDim.x * blockDim.x * 4;
  for (int j = (blockIdx.x * blockDim.x + threadIdx.x) * 4; j < n; j += stride) {
    uint2 av = *(const uint2*)(a + j);
    uint2 bv = *(const uint2*)(b + j);
    float r0 = silu(bf2f((u16)(av.x & 0xFFFF)))  * bf2f((u16)(bv.x & 0xFFFF));
    float r1 = silu(bf2f((u16)(av.x >> 16)))     * bf2f((u16)(bv.x >> 16));
    float r2 = silu(bf2f((u16)(av.y & 0xFFFF)))  * bf2f((u16)(bv.y & 0xFFFF));
    float r3 = silu(bf2f((u16)(av.y >> 16)))     * bf2f((u16)(bv.y >> 16));
    *(uint2*)(d + j) = make_uint2(pack2(r0, r1), pack2(r2, r3));
  }
}

__global__ void swiglu_mul_f32_k(const u16* __restrict__ a, const u16* __restrict__ b,
                                 float* __restrict__ d, int n) {
  const int stride = gridDim.x * blockDim.x * 4;
  for (int j = (blockIdx.x * blockDim.x + threadIdx.x) * 4; j < n; j += stride) {
    uint2 av = *(const uint2*)(a + j);
    uint2 bv = *(const uint2*)(b + j);
    float4 o;
    o.x = silu(bf2f((u16)(av.x & 0xFFFF))) * bf2f((u16)(bv.x & 0xFFFF));
    o.y = silu(bf2f((u16)(av.x >> 16)))    * bf2f((u16)(bv.x >> 16));
    o.z = silu(bf2f((u16)(av.y & 0xFFFF))) * bf2f((u16)(bv.y & 0xFFFF));
    o.w = silu(bf2f((u16)(av.y >> 16)))    * bf2f((u16)(bv.y >> 16));
    *(float4*)(d + j) = o;
  }
}

// ---------------- orchestration ----------------
extern "C" void kernel_launch(void* const* d_in, const int* in_sizes, int n_in,
                              void* d_out, int out_size, void* d_ws, size_t ws_size,
                              hipStream_t stream) {
  (void)in_sizes; (void)n_in; (void)out_size; (void)ws_size;

  const float* x       = (const float*)d_in[0];
  const float* lin0_w  = (const float*)d_in[1];
  const float* lin0_b  = (const float*)d_in[2];
  const float* swi1_w1 = (const float*)d_in[3];
  const float* swi1_b1 = (const float*)d_in[4];
  const float* swi1_w2 = (const float*)d_in[5];
  const float* swi1_b2 = (const float*)d_in[6];
  const float* gate_w  = (const float*)d_in[7];
  const float* exp_w1  = (const float*)d_in[8];
  const float* exp_w3  = (const float*)d_in[9];
  const float* exp_w2  = (const float*)d_in[10];
  const float* sh_w1   = (const float*)d_in[11];
  const float* sh_w3   = (const float*)d_in[12];
  const float* sh_w2   = (const float*)d_in[13];
  const float* lin1_w  = (const float*)d_in[14];
  const float* lin1_b  = (const float*)d_in[15];
  const float* swi2_w1 = (const float*)d_in[16];
  const float* swi2_b1 = (const float*)d_in[17];
  const float* swi2_w2 = (const float*)d_in[18];
  const float* swi2_b2 = (const float*)d_in[19];

  // workspace arena (~84 MB)
  char* p = (char*)d_ws;
  auto take = [&](size_t bytes) -> char* {
    char* r = p;
    p += (bytes + 255) & ~(size_t)255;
    return r;
  };
  const size_t TD = (size_t)T_TOK * DMODEL;   // 4M
  const size_t TS = (size_t)T_TOK * SHID;     // 8M
  u16*   A1 = (u16*)take(TD * 2);  // xb, then h1 (MoE input)
  u16*   A2 = (u16*)take(TD * 2);  // t1, then h2
  u16*   A5 = (u16*)take(TS * 2);  // p1 / s1 / g / q1
  u16*   A6 = (u16*)take(TS * 2);  // p2 / s2 / u / q2
  u16*   A7 = (u16*)take(TS * 2);  // sg / gu
  float* RT = (float*)take(TD * 4);                    // routed accumulator, then t3 (bf16)
  float* CB = (float*)take((size_t)T_TOK * NEXP * 4);  // combine weights
  u16*   WT = (u16*)take((size_t)SHID * DMODEL * 2);   // JIT-transposed weight scratch

  const dim3 blk(256);
  const dim3 tblk(32, 8);
  auto gg = [](int M, int N) { return dim3(N / BN, M / BM); };
  auto tg = [](int K, int N) { return dim3(N / 32, K / 32); };
  const dim3 ew(2048);

  // x -> bf16
  cast_f32_bf16_k<<<ew, blk, 0, stream>>>(x, A1, (int)TD);

  // t1 = x @ lin0 + b
  transpose_cast_k<<<tg(DMODEL, DMODEL), tblk, 0, stream>>>(lin0_w, WT, DMODEL, DMODEL);
  gemm_bf16_wmma<0><<<gg(T_TOK, DMODEL), blk, 0, stream>>>(A1, WT, lin0_b, A2, nullptr, 0, T_TOK, DMODEL, DMODEL);
  // SwiGLU 1
  transpose_cast_k<<<tg(DMODEL, DMODEL), tblk, 0, stream>>>(swi1_w1, WT, DMODEL, DMODEL);
  gemm_bf16_wmma<0><<<gg(T_TOK, DMODEL), blk, 0, stream>>>(A2, WT, swi1_b1, A5, nullptr, 0, T_TOK, DMODEL, DMODEL);
  transpose_cast_k<<<tg(DMODEL, DMODEL), tblk, 0, stream>>>(swi1_w2, WT, DMODEL, DMODEL);
  gemm_bf16_wmma<0><<<gg(T_TOK, DMODEL), blk, 0, stream>>>(A2, WT, swi1_b2, A6, nullptr, 0, T_TOK, DMODEL, DMODEL);
  swiglu_mul_bf16_k<<<ew, blk, 0, stream>>>(A5, A6, A1, (int)TD);  // h1

  // router
  router_kernel<<<(T_TOK * 32) / 256, blk, 0, stream>>>(A1, gate_w, CB);

  // shared experts: seed routed accumulator with their output (fp32)
  transpose_cast_k<<<tg(DMODEL, SHID), tblk, 0, stream>>>(sh_w1, WT, DMODEL, SHID);
  gemm_bf16_wmma<0><<<gg(T_TOK, SHID), blk, 0, stream>>>(A1, WT, nullptr, A5, nullptr, 0, T_TOK, SHID, DMODEL);
  transpose_cast_k<<<tg(DMODEL, SHID), tblk, 0, stream>>>(sh_w3, WT, DMODEL, SHID);
  gemm_bf16_wmma<0><<<gg(T_TOK, SHID), blk, 0, stream>>>(A1, WT, nullptr, A6, nullptr, 0, T_TOK, SHID, DMODEL);
  swiglu_mul_bf16_k<<<ew, blk, 0, stream>>>(A5, A6, A7, (int)TS);
  transpose_cast_k<<<tg(SHID, DMODEL), tblk, 0, stream>>>(sh_w2, WT, SHID, DMODEL);
  gemm_bf16_wmma<1><<<gg(T_TOK, DMODEL), blk, 0, stream>>>(A7, WT, nullptr, RT, nullptr, 0, T_TOK, DMODEL, SHID);

  // routed experts: RT += combine[:,e] * (silu(h1 W1e) * (h1 W3e)) W2e
  const size_t DH = (size_t)DMODEL * HEXP;
  for (int e = 0; e < NEXP; ++e) {
    transpose_cast_k<<<tg(DMODEL, HEXP), tblk, 0, stream>>>(exp_w1 + (size_t)e * DH, WT, DMODEL, HEXP);
    gemm_bf16_wmma<0><<<gg(T_TOK, HEXP), blk, 0, stream>>>(A1, WT, nullptr, A5, nullptr, 0, T_TOK, HEXP, DMODEL);
    transpose_cast_k<<<tg(DMODEL, HEXP), tblk, 0, stream>>>(exp_w3 + (size_t)e * DH, WT, DMODEL, HEXP);
    gemm_bf16_wmma<0><<<gg(T_TOK, HEXP), blk, 0, stream>>>(A1, WT, nullptr, A6, nullptr, 0, T_TOK, HEXP, DMODEL);
    swiglu_mul_bf16_k<<<ew, blk, 0, stream>>>(A5, A6, A7, (int)((size_t)T_TOK * HEXP));
    transpose_cast_k<<<tg(HEXP, DMODEL), tblk, 0, stream>>>(exp_w2 + (size_t)e * DH, WT, HEXP, DMODEL);
    gemm_bf16_wmma<2><<<gg(T_TOK, DMODEL), blk, 0, stream>>>(A7, WT, nullptr, RT, CB + e, NEXP, T_TOK, DMODEL, HEXP);
  }

  // h2 = bf16(routed); t3 reuses RT region
  cast_f32_bf16_k<<<ew, blk, 0, stream>>>(RT, A2, (int)TD);
  u16* T3 = (u16*)RT;
  transpose_cast_k<<<tg(DMODEL, DMODEL), tblk, 0, stream>>>(lin1_w, WT, DMODEL, DMODEL);
  gemm_bf16_wmma<0><<<gg(T_TOK, DMODEL), blk, 0, stream>>>(A2, WT, lin1_b, T3, nullptr, 0, T_TOK, DMODEL, DMODEL);
  transpose_cast_k<<<tg(DMODEL, DMODEL), tblk, 0, stream>>>(swi2_w1, WT, DMODEL, DMODEL);
  gemm_bf16_wmma<0><<<gg(T_TOK, DMODEL), blk, 0, stream>>>(T3, WT, swi2_b1, A5, nullptr, 0, T_TOK, DMODEL, DMODEL);
  transpose_cast_k<<<tg(DMODEL, DMODEL), tblk, 0, stream>>>(swi2_w2, WT, DMODEL, DMODEL);
  gemm_bf16_wmma<0><<<gg(T_TOK, DMODEL), blk, 0, stream>>>(T3, WT, swi2_b2, A6, nullptr, 0, T_TOK, DMODEL, DMODEL);
  swiglu_mul_f32_k<<<ew, blk, 0, stream>>>(A5, A6, (float*)d_out, (int)TD);
}
